// Nerf2D_1262720385270
// MI455X (gfx1250) — compile-verified, matrix operand order checked by reading, not run on hardware
//
#include <hip/hip_runtime.h>
#include <cstdint>

typedef __attribute__((ext_vector_type(16))) _Float16 v16h;
typedef __attribute__((ext_vector_type(8)))  float    v8f;

#define WMMA_F16F32(A, B, C) \
  __builtin_amdgcn_wmma_f32_16x16x32_f16(false, (A), false, (B), (short)0, (C), false, false)

#define N_ENC    65536
#define N_LEVELS 16
#define N_HID    64
#define TS       70          // tile row stride in halves (bank-conflict-free fragment loads)

// RES[l] = floor(16 * (512/16)^(l/15)) computed exactly as the reference does
__constant__ int RES_TAB[N_LEVELS] = {16, 20, 25, 32, 40, 50, 64, 80,
                                      101, 128, 161, 203, 256, 322, 406, 512};

// Load one 16x32 f16 operand fragment (A or B role) from an operand-major LDS
// tile: element (rc, k) lives at base[rc*TS + k].  CDNA5 16-bit operand layout:
//   lanes 0-15 : rc = lane,     K = {0..7, 16..23}   (pairs per VGPR)
//   lanes 16-31: rc = lane-16,  K = {8..15, 24..31}
__device__ __forceinline__ v16h load_frag(const _Float16* base, int rc, int h16, int kofs) {
  const uint32_t* q = (const uint32_t*)(base + rc * TS + kofs + h16 * 8);
  v16h f;
  uint32_t* fu = (uint32_t*)&f;
#pragma unroll
  for (int g = 0; g < 8; ++g) {
    // g<4 -> k = 2g (+h16*8); g>=4 -> k = 16 + 2(g-4) (+h16*8)
    fu[g] = (g < 4) ? q[g] : q[g + 4];
  }
  return f;
}

__global__ __launch_bounds__(256)
void nerf2d_fused(const float2* __restrict__ xn,
                  const float2* __restrict__ emb,     // (16, 65536, 2) as float2
                  const float*  __restrict__ W1,      // (32,64)
                  const float*  __restrict__ b1,      // (64)
                  const float*  __restrict__ W2,      // (64,64)
                  const float*  __restrict__ b2,      // (64)
                  const float*  __restrict__ W3,      // (64,3)
                  const float*  __restrict__ b3,      // (3)
                  float*        __restrict__ out,     // (B,3)
                  int nPoints) {
  __shared__ _Float16 sW1[N_HID * TS];     // [n][k], k<32 valid
  __shared__ _Float16 sW2[N_HID * TS];     // [n][k], k<64 valid
  __shared__ _Float16 sW3[16 * TS];        // [n][k], n>=3 zero-padded
  __shared__ float    sB1[N_HID], sB2[N_HID], sB3[16];
  __shared__ _Float16 sTile[8 * 32 * TS];  // per-wave 32-row activation tile

  const int tid  = threadIdx.x;
  const int wave = tid >> 5;
  const int lane = tid & 31;
  const int h16  = lane >> 4;   // half-wave select
  const int rc   = lane & 15;   // row (A) / col (B,C) index

  // ---- cooperative weight conversion to f16, operand-major ----
  for (int i = tid; i < 32 * N_HID; i += 256) {
    int k = i & 31, n = i >> 5;
    sW1[n * TS + k] = (_Float16)W1[k * N_HID + n];
  }
  for (int i = tid; i < N_HID * N_HID; i += 256) {
    int k = i & 63, n = i >> 6;
    sW2[n * TS + k] = (_Float16)W2[k * N_HID + n];
  }
  for (int i = tid; i < N_HID * 16; i += 256) {
    int k = i & 63, n = i >> 6;
    sW3[n * TS + k] = (n < 3) ? (_Float16)W3[k * 3 + n] : (_Float16)0.0f;
  }
  if (tid < N_HID) { sB1[tid] = b1[tid]; sB2[tid] = b2[tid]; }
  if (tid < 16)    { sB3[tid] = (tid < 3) ? b3[tid] : 0.0f; }
  __syncthreads();

  _Float16* myTile = sTile + wave * 32 * TS;
  const int  waveBase = blockIdx.x * 256 + wave * 32;

  // ---- hash-grid encode: one point per lane, 32 f16 features -> LDS row ----
  {
    const int pt = waveBase + lane;
    float2 xy = (pt < nPoints) ? xn[pt] : make_float2(0.f, 0.f);
    const float x01 = (xy.x + 1.0f) * 0.5f;
    const float y01 = (xy.y + 1.0f) * 0.5f;
    uint32_t* rowU = (uint32_t*)(myTile + lane * TS);
#pragma unroll
    for (int l = 0; l < N_LEVELS; ++l) {
      const int res = RES_TAB[l];
      const float rf = (float)(res - 1);
      const float xf = x01 * rf, yf = y01 * rf;
      const float fx = floorf(xf), fy = floorf(yf);
      const float wx = xf - fx,  wy = yf - fy;
      int c0x = (int)fx; c0x = min(max(c0x, 0), res - 1);
      int c0y = (int)fy; c0y = min(max(c0y, 0), res - 1);
      const int c1x = min(c0x + 1, res - 1);
      const int c1y = min(c0y + 1, res - 1);
      uint32_t i00, i10, i01, i11;
      if (res <= 256) {               // res*res <= N_ENC : direct index
        i00 = (uint32_t)(c0y * res + c0x);
        i10 = (uint32_t)(c0y * res + c1x);
        i01 = (uint32_t)(c1y * res + c0x);
        i11 = (uint32_t)(c1y * res + c1x);
      } else {                        // spatial hash
        const uint32_t P = 2654435761u;
        i00 = ((uint32_t)c0x ^ ((uint32_t)c0y * P)) & 65535u;
        i10 = ((uint32_t)c1x ^ ((uint32_t)c0y * P)) & 65535u;
        i01 = ((uint32_t)c0x ^ ((uint32_t)c1y * P)) & 65535u;
        i11 = ((uint32_t)c1x ^ ((uint32_t)c1y * P)) & 65535u;
      }
      const float2* tab = emb + l * N_ENC;
      const float2 g00 = tab[i00], g10 = tab[i10], g01 = tab[i01], g11 = tab[i11];
      const float w00 = (1.f - wx) * (1.f - wy);
      const float w10 = wx * (1.f - wy);
      const float w01 = (1.f - wx) * wy;
      const float w11 = wx * wy;
      const float f0 = w00 * g00.x + w10 * g10.x + w01 * g01.x + w11 * g11.x;
      const float f1 = w00 * g00.y + w10 * g10.y + w01 * g01.y + w11 * g11.y;
      union { _Float16 h[2]; uint32_t u; } pk;
      pk.h[0] = (_Float16)f0; pk.h[1] = (_Float16)f1;
      rowU[l] = pk.u;                 // features 2l, 2l+1 packed
    }
  }
  __builtin_amdgcn_wave_barrier();    // DS in-order per wave; block reordering

  // ---- layer 1: H1 = relu(F @ W1 + b1),  F:32x32  W1:32x64 ----
  v16h a1_0 = load_frag(myTile,           rc, h16, 0);
  v16h a1_1 = load_frag(myTile + 16 * TS, rc, h16, 0);
#pragma unroll
  for (int jt = 0; jt < 4; ++jt) {
    const v16h bf = load_frag(sW1, jt * 16 + rc, h16, 0);
    v8f acc0 = {}, acc1 = {};
    acc0 = WMMA_F16F32(a1_0, bf, acc0);
    acc1 = WMMA_F16F32(a1_1, bf, acc1);
    const float bias = sB1[jt * 16 + rc];
    const int col = jt * 16 + rc;
#pragma unroll
    for (int r = 0; r < 8; ++r) {
      float v0 = acc0[r] + bias; v0 = v0 > 0.f ? v0 : 0.f;
      float v1 = acc1[r] + bias; v1 = v1 > 0.f ? v1 : 0.f;
      myTile[(r + 8 * h16) * TS + col]        = (_Float16)v0;  // overwrites feats (already in regs)
      myTile[(16 + r + 8 * h16) * TS + col]   = (_Float16)v1;
    }
  }
  __builtin_amdgcn_wave_barrier();

  // ---- layer 2: H2 = relu(H1 @ W2 + b2),  K=64 -> two WMMAs per tile ----
  v16h a2[2][2];
  a2[0][0] = load_frag(myTile,           rc, h16, 0);
  a2[0][1] = load_frag(myTile,           rc, h16, 32);
  a2[1][0] = load_frag(myTile + 16 * TS, rc, h16, 0);
  a2[1][1] = load_frag(myTile + 16 * TS, rc, h16, 32);
  __builtin_amdgcn_wave_barrier();
#pragma unroll
  for (int jt = 0; jt < 4; ++jt) {
    const v16h bk0 = load_frag(sW2, jt * 16 + rc, h16, 0);
    const v16h bk1 = load_frag(sW2, jt * 16 + rc, h16, 32);
    const float bias = sB2[jt * 16 + rc];
    const int col = jt * 16 + rc;
#pragma unroll
    for (int mt = 0; mt < 2; ++mt) {
      v8f acc = {};
      acc = WMMA_F16F32(a2[mt][0], bk0, acc);
      acc = WMMA_F16F32(a2[mt][1], bk1, acc);
#pragma unroll
      for (int r = 0; r < 8; ++r) {
        float v = acc[r] + bias; v = v > 0.f ? v : 0.f;
        myTile[(mt * 16 + r + 8 * h16) * TS + col] = (_Float16)v;
      }
    }
  }
  __builtin_amdgcn_wave_barrier();

  // ---- layer 3: out = sigmoid(H2 @ W3 + b3), W3 zero-padded to 64x16 ----
  v16h a3[2][2];
  a3[0][0] = load_frag(myTile,           rc, h16, 0);
  a3[0][1] = load_frag(myTile,           rc, h16, 32);
  a3[1][0] = load_frag(myTile + 16 * TS, rc, h16, 0);
  a3[1][1] = load_frag(myTile + 16 * TS, rc, h16, 32);
  const v16h c3k0 = load_frag(sW3, rc, h16, 0);
  const v16h c3k1 = load_frag(sW3, rc, h16, 32);
  const float bias3 = sB3[rc];
#pragma unroll
  for (int mt = 0; mt < 2; ++mt) {
    v8f acc = {};
    acc = WMMA_F16F32(a3[mt][0], c3k0, acc);
    acc = WMMA_F16F32(a3[mt][1], c3k1, acc);
#pragma unroll
    for (int r = 0; r < 8; ++r) {
      const float z = acc[r] + bias3;
      const float s = 1.0f / (1.0f + __expf(-z));
      const int point = waveBase + mt * 16 + r + 8 * h16;
      if (rc < 3 && point < nPoints) out[point * 3 + rc] = s;
    }
  }
}

extern "C" void kernel_launch(void* const* d_in, const int* in_sizes, int n_in,
                              void* d_out, int out_size, void* d_ws, size_t ws_size,
                              hipStream_t stream) {
  const float2* xn  = (const float2*)d_in[0];
  const float2* emb = (const float2*)d_in[1];   // (16,65536,2) f32
  const float*  W1  = (const float*)d_in[2];
  const float*  b1  = (const float*)d_in[3];
  const float*  W2  = (const float*)d_in[4];
  const float*  b2  = (const float*)d_in[5];
  const float*  W3  = (const float*)d_in[6];
  const float*  b3  = (const float*)d_in[7];
  float* out = (float*)d_out;

  const int nPoints = in_sizes[0] / 2;          // xn is (B,2)
  const int grid = (nPoints + 255) / 256;       // B=2^21 -> exact multiple
  nerf2d_fused<<<grid, 256, 0, stream>>>(xn, emb, W1, b1, W2, b2, W3, b3, out, nPoints);
}
